// EdgeHGTConv_81372450390265
// MI455X (gfx1250) — matrix operand order repaired and checked by reading, not compiled, not run on hardware
//
#include <hip/hip_runtime.h>
#include <hip/hip_bf16.h>

typedef _Float16 f16;
typedef f16   v16h __attribute__((ext_vector_type(16)));
typedef float v8f  __attribute__((ext_vector_type(8)));

#define TN_ 4
#define TE_ 8
#define H_  4
#define HS_ 16

// meta layout (ints): [0..7] counts, [8..16] offsets, [17..24] cursors, [25..33] tileOffsets
#define CB 0
#define OB 8
#define UB 17
#define TB 25

// ---------- float <-> monotonic uint key for atomicMax on floats ----------
__device__ inline unsigned fkey(float f) {
    unsigned u = __float_as_uint(f);
    return (u & 0x80000000u) ? ~u : (u | 0x80000000u);
}
__device__ inline float fdec(unsigned k) {
    return (k & 0x80000000u) ? __uint_as_float(k & 0x7fffffffu) : __uint_as_float(~k);
}

__device__ inline v8f wmma_f16(v16h a, v16h b, v8f c) {
    return __builtin_amdgcn_wmma_f32_16x16x32_f16(false, a, false, b, (short)0, c, false, false);
}

// ---------------------------------------------------------------- fills ----
__global__ void fill_kernel(unsigned* p, unsigned v, size_t n) {
    size_t i = (size_t)blockIdx.x * blockDim.x + threadIdx.x;
    if (i < n) p[i] = v;
}

// ------------------------------------------------------- edge bucketing ----
__global__ void count_kernel(const int* __restrict__ et, int* meta, int E) {
    int i = blockIdx.x * blockDim.x + threadIdx.x;
    if (i < E) atomicAdd(&meta[CB + et[i]], 1);
}
__global__ void prefix_kernel(int* meta) {
    if (threadIdx.x == 0) {
        meta[OB] = 0;
        for (int t = 0; t < TE_; ++t) {
            meta[OB + t + 1] = meta[OB + t] + meta[CB + t];
            meta[UB + t]     = meta[OB + t];
        }
        int tt = 0;
        meta[TB] = 0;
        for (int t = 0; t < TE_; ++t) {
            tt += (meta[CB + t] + 15) >> 4;
            meta[TB + t + 1] = tt;
        }
    }
}
__global__ void scatter_kernel(const int* __restrict__ et, int* meta, int* perm, int E) {
    int i = blockIdx.x * blockDim.x + threadIdx.x;
    if (i < E) {
        int t = et[i];
        int p = atomicAdd(&meta[UB + t], 1);
        perm[p] = i;
    }
}

// ----------------------------------------- weight pre-swizzle to f16 B -----
// B operand, this code's contiguous k order: lane n<16 -> column N=n, halfs j -> K=j;
// lane n>=16 -> column N=n-16, halfs j -> K=16+j.
__global__ void prep_att(const float* __restrict__ W, f16* __restrict__ out) {
    int idx = blockIdx.x * blockDim.x + threadIdx.x;
    if (idx >= TE_ * H_ * 16 * 2 * 32 * 16) return;
    int j    = idx & 15;
    int lane = (idx >> 4) & 31;
    int ks   = (idx >> 9) & 1;
    int d    = (idx >> 10) & 15;
    int h    = (idx >> 14) & 3;
    int t    = idx >> 16;
    int col  = lane & 15;
    int K    = ((lane >> 4) << 4) + j;
    int i    = ks * 32 + K;                 // EIN index
    int o    = h * 256 + d * 16 + col;      // output feature
    out[idx] = (f16)W[((size_t)t * 64 + i) * 1024 + o];
}
__global__ void prep_msg(const float* __restrict__ W, f16* __restrict__ out) {
    int idx = blockIdx.x * blockDim.x + threadIdx.x;
    if (idx >= TE_ * H_ * 2 * 32 * 16) return;
    int j    = idx & 15;
    int lane = (idx >> 4) & 31;
    int ks   = (idx >> 9) & 1;
    int nt   = (idx >> 10) & 3;             // head block of the 64 cols
    int t    = idx >> 12;
    int col  = lane & 15;
    int K    = ((lane >> 4) << 4) + j;
    int i    = ks * 32 + K;
    int o    = nt * 16 + col;
    out[idx] = (f16)W[((size_t)t * 64 + i) * 64 + o];
}
// Wra / Wrm: 16x16, zero-padded to K=32
__global__ void prep_rel(const float* __restrict__ W, f16* __restrict__ out) {
    int idx = blockIdx.x * blockDim.x + threadIdx.x;
    if (idx >= H_ * TE_ * 32 * 16) return;
    int j    = idx & 15;
    int lane = (idx >> 4) & 31;
    int t    = (idx >> 9) & 7;
    int h    = idx >> 12;
    int col  = lane & 15;
    int K    = ((lane >> 4) << 4) + j;      // d index
    out[idx] = (K < 16) ? (f16)W[(((size_t)h * TE_ + t) * 16 + K) * 16 + col] : (f16)0;
}

// ---------------------------------------------------- typed node linears ---
__global__ __launch_bounds__(256) void node_qkv_kernel(
    const float* __restrict__ x, const int* __restrict__ ntype,
    const float* __restrict__ Wk, const float* __restrict__ Wq, const float* __restrict__ Wv,
    float* __restrict__ kb, float* __restrict__ qb, float* __restrict__ vb, int N)
{
    int warp = (blockIdx.x * blockDim.x + threadIdx.x) >> 5;
    if (warp >= N) return;
    int lane = threadIdx.x & 31;
    const float* xr = x + (size_t)warp * 64;
    float x0 = xr[lane], x1 = xr[lane + 32];
    int t = ntype[warp];
    const float* wk = Wk + (size_t)t * 4096;
    const float* wq = Wq + (size_t)t * 4096;
    const float* wv = Wv + (size_t)t * 4096;
    float ak0 = 0, ak1 = 0, aq0 = 0, aq1 = 0, av0 = 0, av1 = 0;
    for (int i = 0; i < 64; ++i) {
        float xi = (i < 32) ? __shfl(x0, i) : __shfl(x1, i - 32);
        float2 wk2 = ((const float2*)(wk + (size_t)i * 64))[lane];
        float2 wq2 = ((const float2*)(wq + (size_t)i * 64))[lane];
        float2 wv2 = ((const float2*)(wv + (size_t)i * 64))[lane];
        ak0 += xi * wk2.x; ak1 += xi * wk2.y;
        aq0 += xi * wq2.x; aq1 += xi * wq2.y;
        av0 += xi * wv2.x; av1 += xi * wv2.y;
    }
    size_t o = (size_t)warp * 64 + lane * 2;
    kb[o] = ak0; kb[o + 1] = ak1;
    qb[o] = aq0; qb[o + 1] = aq1;
    vb[o] = av0; vb[o + 1] = av1;
}

// --------------------------------------------------- the big edge kernel ---
// 1 block = 128 threads = 4 waves; wave h handles head h for a tile of 16
// same-etype edges.  WMMA:  edge_att = x_edge @ We_att   (fused with k-dot)
//                           kw  = k_e @ Wra  (+ Sum_d k_d * edge_att[d,:])
//                           msg = v_e @ Wrm  +  x_edge @ We_msg
__global__ __launch_bounds__(128) void edge_attn_kernel(
    const float* __restrict__ xe,
    const float* __restrict__ kb, const float* __restrict__ qb, const float* __restrict__ vb,
    const int* __restrict__ src, const int* __restrict__ dst,
    const int* __restrict__ perm, const int* __restrict__ meta,
    const f16* __restrict__ WeAttSw, const f16* __restrict__ WeMsgSw,
    const f16* __restrict__ WraSw, const f16* __restrict__ WrmSw,
    const float* __restrict__ pri,
    float* __restrict__ abuf, float* __restrict__ msgbuf, unsigned* __restrict__ amax)
{
    __shared__ int   s_e[16], s_srcn[16], s_dstn[16];
    __shared__ f16   s_x[16][64];
    __shared__ float s_k[16][64], s_q[16][64], s_v[16][64];

    int bid = blockIdx.x;
    int ntiles = meta[TB + TE_];
    if (bid >= ntiles) return;
    int t = 0;
    while (t < TE_ - 1 && bid >= meta[TB + t + 1]) ++t;
    int ltile = bid - meta[TB + t];
    int cnt   = meta[CB + t];
    int base  = meta[OB + t] + ltile * 16;
    int m     = cnt - ltile * 16; if (m > 16) m = 16;

    int tid = threadIdx.x;
    if (tid < 16) {
        int e = perm[base + (tid < m ? tid : 0)];
        s_e[tid] = e; s_srcn[tid] = src[e]; s_dstn[tid] = dst[e];
    }
    __syncthreads();
    for (int i = tid; i < 16 * 64; i += 128) {
        int r = i >> 6, c = i & 63;
        s_x[r][c] = (f16)xe[(size_t)s_e[r] * 64 + c];
        s_k[r][c] = kb[(size_t)s_srcn[r] * 64 + c];
        s_q[r][c] = qb[(size_t)s_dstn[r] * 64 + c];
        s_v[r][c] = vb[(size_t)s_srcn[r] * 64 + c];
    }
    __syncthreads();

    int h    = tid >> 5;
    int lane = tid & 31;
    int rowA = lane & 15;
    int hs   = lane >> 4;   // half-select

    // A operands (16-bit A layout, contiguous: element j -> K=16*hs+j)
    v16h ax0, ax1, akk, avv;
    #pragma unroll
    for (int j = 0; j < 16; ++j) { akk[j] = (f16)0; avv[j] = (f16)0; }
    #pragma unroll
    for (int p = 0; p < 8; ++p) {
        int K0 = 2 * p + (hs << 4);
        ax0[2 * p]     = s_x[rowA][K0];
        ax0[2 * p + 1] = s_x[rowA][K0 + 1];
        ax1[2 * p]     = s_x[rowA][32 + K0];
        ax1[2 * p + 1] = s_x[rowA][32 + K0 + 1];
    }
    #pragma unroll
    for (int p = 0; p < 8; ++p) {
        int K0 = 2 * p;
        akk[2 * p]     = (f16)s_k[rowA][h * 16 + K0];
        akk[2 * p + 1] = (f16)s_k[rowA][h * 16 + K0 + 1];
        avv[2 * p]     = (f16)s_v[rowA][h * 16 + K0];
        avv[2 * p + 1] = (f16)s_v[rowA][h * 16 + K0 + 1];
    }

    // kw = k_e @ Wra[h][t]
    v8f kw;
    #pragma unroll
    for (int r = 0; r < 8; ++r) kw[r] = 0.f;
    {
        v16h bw = *(const v16h*)(WraSw + (((size_t)h * TE_ + t) * 32 + lane) * 16);
        kw = wmma_f16(akk, bw, kw);
    }

    // fused: per d, edge_att 16x16 tile via WMMA, then kw += k_d * tile
    const f16* pb = WeAttSw + (((size_t)t * H_ + h) * 16) * 2 * 32 * 16;
    #pragma unroll 1
    for (int d = 0; d < 16; ++d) {
        v8f dt;
        #pragma unroll
        for (int r = 0; r < 8; ++r) dt[r] = 0.f;
        v16h b0 = *(const v16h*)(pb + (((size_t)d * 2 + 0) * 32 + lane) * 16);
        v16h b1 = *(const v16h*)(pb + (((size_t)d * 2 + 1) * 32 + lane) * 16);
        dt = wmma_f16(ax0, b0, dt);
        dt = wmma_f16(ax1, b1, dt);
        #pragma unroll
        for (int r = 0; r < 8; ++r)
            kw[r] += s_k[r + (hs << 3)][h * 16 + d] * dt[r];
    }

    // a[e,h] = sum_f kw*q  * pri/sqrt(HS)
    float scale = pri[h * TE_ + t] * 0.25f;
    float red[8];
    #pragma unroll
    for (int r = 0; r < 8; ++r) {
        float s = kw[r] * s_q[r + (hs << 3)][h * 16 + (lane & 15)];
        s += __shfl_xor(s, 1);
        s += __shfl_xor(s, 2);
        s += __shfl_xor(s, 4);
        s += __shfl_xor(s, 8);
        red[r] = s * scale;
    }
    if ((lane & 15) == 0) {
        #pragma unroll
        for (int r = 0; r < 8; ++r) {
            int row = r + (hs << 3);
            if (row < m) {
                abuf[(size_t)s_e[row] * H_ + h] = red[r];
                atomicMax(&amax[(size_t)s_dstn[row] * H_ + h], fkey(red[r]));
            }
        }
    }

    // msg = v_e @ Wrm + x_edge @ We_msg
    v8f mg;
    #pragma unroll
    for (int r = 0; r < 8; ++r) mg[r] = 0.f;
    {
        v16h bw = *(const v16h*)(WrmSw + (((size_t)h * TE_ + t) * 32 + lane) * 16);
        mg = wmma_f16(avv, bw, mg);
        const f16* pm = WeMsgSw + ((size_t)t * H_ + h) * 2 * 32 * 16;
        v16h b0 = *(const v16h*)(pm + ((size_t)lane) * 16);
        v16h b1 = *(const v16h*)(pm + ((size_t)32 + lane) * 16);
        mg = wmma_f16(ax0, b0, mg);
        mg = wmma_f16(ax1, b1, mg);
    }
    #pragma unroll
    for (int r = 0; r < 8; ++r) {
        int row = r + (hs << 3);
        if (row < m)
            msgbuf[(size_t)s_e[row] * 64 + h * 16 + (lane & 15)] = mg[r];
    }
}

// ----------------------------------------------------------- softmax -------
__global__ void edge_exp_kernel(const int* __restrict__ dst, float* __restrict__ abuf,
                                const unsigned* __restrict__ amax, float* __restrict__ den, int E)
{
    int idx = blockIdx.x * blockDim.x + threadIdx.x;
    if (idx >= E * H_) return;
    int e = idx >> 2, h = idx & 3;
    int d = dst[e];
    float mx = fdec(amax[(size_t)d * H_ + h]);
    float ex = __expf(abuf[idx] - mx);
    abuf[idx] = ex;
    atomicAdd(&den[(size_t)d * H_ + h], ex);
}

__global__ void edge_scatter_kernel(const int* __restrict__ dst,
                                    const float* __restrict__ abuf, const float* __restrict__ den,
                                    const float* __restrict__ msgbuf, float* __restrict__ out, int E)
{
    int idx = blockIdx.x * blockDim.x + threadIdx.x;
    if (idx >= E * 64) return;
    int e = idx >> 6, o = idx & 63, h = o >> 4;
    int d = dst[e];
    float att = abuf[(size_t)e * H_ + h] / den[(size_t)d * H_ + h];
    atomicAdd(&out[(size_t)d * 64 + o], msgbuf[idx] * att);
}

// --------------------------------------------- output linear + residual ----
__global__ __launch_bounds__(256) void node_out_kernel(
    const float* __restrict__ x, const int* __restrict__ ntype,
    const float* __restrict__ Wa, const float* __restrict__ skip,
    float* __restrict__ out, int N)
{
    int warp = (blockIdx.x * blockDim.x + threadIdx.x) >> 5;
    if (warp >= N) return;
    int lane = threadIdx.x & 31;
    float h0 = out[(size_t)warp * 64 + lane];
    float h1 = out[(size_t)warp * 64 + lane + 32];
    int t = ntype[warp];
    const float* wa = Wa + (size_t)t * 4096;
    float g0 = 0, g1 = 0;
    for (int i = 0; i < 64; ++i) {
        float hi = (i < 32) ? __shfl(h0, i) : __shfl(h1, i - 32);
        float2 w2 = ((const float2*)(wa + (size_t)i * 64))[lane];
        g0 += hi * w2.x; g1 += hi * w2.y;
    }
    float alpha = 1.f / (1.f + __expf(-skip[t]));
    size_t o = (size_t)warp * 64 + lane * 2;
    float xn0 = x[o], xn1 = x[o + 1];
    out[o]     = g0 * alpha + xn0 * (1.f - alpha);
    out[o + 1] = g1 * alpha + xn1 * (1.f - alpha);
}

// ---------------------------------------------------------------------------
extern "C" void kernel_launch(void* const* d_in, const int* in_sizes, int n_in,
                              void* d_out, int out_size, void* d_ws, size_t ws_size,
                              hipStream_t stream)
{
    const float* x_node = (const float*)d_in[0];
    const float* x_edge = (const float*)d_in[1];
    const int*   ntype  = (const int*)d_in[2];
    const int*   etype  = (const int*)d_in[3];
    const int*   src    = (const int*)d_in[4];
    const int*   dst    = (const int*)d_in[5];
    const float* Wk     = (const float*)d_in[6];
    const float* Wq     = (const float*)d_in[7];
    const float* Wv     = (const float*)d_in[8];
    const float* Wa     = (const float*)d_in[9];
    const float* Wra    = (const float*)d_in[10];
    const float* Wrm    = (const float*)d_in[11];
    const float* WeAtt  = (const float*)d_in[12];
    const float* WeMsg  = (const float*)d_in[13];
    const float* pri    = (const float*)d_in[14];
    const float* skip   = (const float*)d_in[15];

    const int N = in_sizes[0] / 64;
    const int E = in_sizes[1] / 64;

    char* ws = (char*)d_ws;
    size_t off = 0;
    auto alloc = [&](size_t bytes) -> char* {
        char* p = ws + off;
        off = (off + bytes + 255) & ~(size_t)255;
        return p;
    };
    float*    kb      = (float*)alloc((size_t)N * 64 * 4);
    float*    qb      = (float*)alloc((size_t)N * 64 * 4);
    float*    vb      = (float*)alloc((size_t)N * 64 * 4);
    float*    abuf    = (float*)alloc((size_t)E * H_ * 4);
    float*    msgbuf  = (float*)alloc((size_t)E * 64 * 4);
    unsigned* amax    = (unsigned*)alloc((size_t)N * H_ * 4);
    float*    den     = (float*)alloc((size_t)N * H_ * 4);
    int*      perm    = (int*)alloc((size_t)E * 4);
    int*      meta    = (int*)alloc(64 * 4);
    f16*      WeAttSw = (f16*)alloc((size_t)TE_ * H_ * 16 * 2 * 32 * 16 * 2);
    f16*      WeMsgSw = (f16*)alloc((size_t)TE_ * H_ * 2 * 32 * 16 * 2);
    f16*      WraSw   = (f16*)alloc((size_t)H_ * TE_ * 32 * 16 * 2);
    f16*      WrmSw   = (f16*)alloc((size_t)H_ * TE_ * 32 * 16 * 2);

    // zero init: meta, amax(key 0 == -inf class), den, d_out accumulator
    fill_kernel<<<1, 64, 0, stream>>>((unsigned*)meta, 0u, 64);
    fill_kernel<<<(N * H_ + 255) / 256, 256, 0, stream>>>(amax, 0u, (size_t)N * H_);
    fill_kernel<<<(N * H_ + 255) / 256, 256, 0, stream>>>((unsigned*)den, 0u, (size_t)N * H_);
    fill_kernel<<<(N * 64 + 255) / 256, 256, 0, stream>>>((unsigned*)d_out, 0u, (size_t)N * 64);

    // bucket edges by etype
    count_kernel<<<(E + 255) / 256, 256, 0, stream>>>(etype, meta, E);
    prefix_kernel<<<1, 32, 0, stream>>>(meta);
    scatter_kernel<<<(E + 255) / 256, 256, 0, stream>>>(etype, meta, perm, E);

    // pre-swizzle weights to f16 WMMA B-operand layout
    {
        int n1 = TE_ * H_ * 16 * 2 * 32 * 16;
        prep_att<<<(n1 + 255) / 256, 256, 0, stream>>>(WeAtt, WeAttSw);
        int n2 = TE_ * H_ * 2 * 32 * 16;
        prep_msg<<<(n2 + 255) / 256, 256, 0, stream>>>(WeMsg, WeMsgSw);
        int n3 = H_ * TE_ * 32 * 16;
        prep_rel<<<(n3 + 255) / 256, 256, 0, stream>>>(Wra, WraSw);
        prep_rel<<<(n3 + 255) / 256, 256, 0, stream>>>(Wrm, WrmSw);
    }

    // typed node linears
    node_qkv_kernel<<<(N + 7) / 8, 256, 0, stream>>>(x_node, ntype, Wk, Wq, Wv, kb, qb, vb, N);

    // heavy fused edge kernel (WMMA)
    int maxTiles = E / 16 + TE_ + 1;
    edge_attn_kernel<<<maxTiles, 128, 0, stream>>>(
        x_edge, kb, qb, vb, src, dst, perm, meta,
        WeAttSw, WeMsgSw, WraSw, WrmSw, pri, abuf, msgbuf, amax);

    // edge softmax + scatter
    edge_exp_kernel<<<(E * H_ + 255) / 256, 256, 0, stream>>>(dst, abuf, amax, den, E);
    edge_scatter_kernel<<<(E * 64 + 255) / 256, 256, 0, stream>>>(dst, abuf, den, msgbuf, (float*)d_out, E);

    // output linear + skip residual
    node_out_kernel<<<(N + 7) / 8, 256, 0, stream>>>(x_node, ntype, Wa, skip, (float*)d_out, N);
}
